// SmAttentionPool_14508399525952
// MI455X (gfx1250) — matrix operand order repaired and checked by reading, not compiled, not run on hardware
//
#include <hip/hip_runtime.h>
#include <cstdint>

typedef __bf16 bf16;
typedef __attribute__((ext_vector_type(16))) __bf16 v16bf;
typedef __attribute__((ext_vector_type(8)))  __bf16 v8bf;
typedef __attribute__((ext_vector_type(4)))  __bf16 v4bf;
typedef __attribute__((ext_vector_type(8)))  float  v8f;

#define B_ 8
#define N_ 2048
#define D_ 512
#define A_ 128
#define BM 128
#define BN 128
#define KT 32
#define LDK 40   // padded LDS K-stride in bf16 elems (80B -> conflict-free frag reads)

// Issue one K-tile (A: 128x32, B^T: 128x32) as async global->LDS DMA.
// Each of 256 threads moves 2x16B for A and 2x16B for B.
// NOTE: inst offset applies to BOTH the global and the LDS address, and the
// 16 staged elements are contiguous in LDS (padding only at row boundaries),
// so offset:16 covers the second half with the same address registers.
#define STAGE_ASYNC(AsDst, BsDst, KTV, LD)                                     \
  {                                                                            \
    const bf16* ga = Ag + (long)lr * (LD) + (KTV) + lk;                        \
    const bf16* gb = Bg + (long)lr * (LD) + (KTV) + lk;                        \
    unsigned la = (unsigned)(uintptr_t)(const void*)&(AsDst)[lr * LDK + lk];   \
    unsigned lb = (unsigned)(uintptr_t)(const void*)&(BsDst)[lr * LDK + lk];   \
    asm volatile(                                                              \
        "global_load_async_to_lds_b128 %0, %2, off\n\t"                        \
        "global_load_async_to_lds_b128 %0, %2, off offset:16\n\t"              \
        "global_load_async_to_lds_b128 %1, %3, off\n\t"                        \
        "global_load_async_to_lds_b128 %1, %3, off offset:16"                  \
        :: "v"(la), "v"(lb), "v"(ga), "v"(gb) : "memory");                     \
  }

#define WAIT_ASYNC() asm volatile("s_wait_asynccnt 0x0" ::: "memory")

// Load WMMA fragments from an LDS tile and run the 2x4 WMMA block.
#define FRAG_WMMA(AsCur, BsCur)                                                \
  {                                                                            \
    v16bf af[2], bfr[4];                                                       \
    _Pragma("unroll")                                                          \
    for (int mi = 0; mi < 2; ++mi) {                                           \
      const int row = wm + mi * 16 + rl;                                       \
      v8bf lo = *reinterpret_cast<const v8bf*>(&(AsCur)[row * LDK + ka]);      \
      v8bf hi = *reinterpret_cast<const v8bf*>(&(AsCur)[row * LDK + ka + 16]); \
      _Pragma("unroll")                                                        \
      for (int i = 0; i < 8; ++i) { af[mi][i] = lo[i]; af[mi][i + 8] = hi[i]; }\
    }                                                                          \
    _Pragma("unroll")                                                          \
    for (int ni = 0; ni < 4; ++ni) {                                           \
      const int col = wn + ni * 16 + rl;                                       \
      v8bf lo = *reinterpret_cast<const v8bf*>(&(BsCur)[col * LDK + kb]);      \
      v8bf hi = *reinterpret_cast<const v8bf*>(&(BsCur)[col * LDK + kb + 16]); \
      _Pragma("unroll")                                                        \
      for (int i = 0; i < 8; ++i) { bfr[ni][i] = lo[i]; bfr[ni][i + 8] = hi[i]; }\
    }                                                                          \
    _Pragma("unroll")                                                          \
    for (int mi = 0; mi < 2; ++mi)                                             \
      _Pragma("unroll")                                                        \
      for (int ni = 0; ni < 4; ++ni)                                           \
        acc[mi][ni] = __builtin_amdgcn_wmma_f32_16x16x32_bf16(                 \
            false, af[mi], false, bfr[ni], (short)0, acc[mi][ni], false, false);\
  }

// ---------------------------------------------------------------- converts
__global__ __launch_bounds__(256) void cvt4_kernel(const float* __restrict__ src,
                                                   bf16* __restrict__ dst, long n4) {
  long i = (long)blockIdx.x * blockDim.x + threadIdx.x;
  if (i >= n4) return;
  const float4 v = reinterpret_cast<const float4*>(src)[i];
  v4bf o; o[0] = (bf16)v.x; o[1] = (bf16)v.y; o[2] = (bf16)v.z; o[3] = (bf16)v.w;
  reinterpret_cast<v4bf*>(dst)[i] = o;
}

// g0_t[b][d][n] = bf16(X[b][n][d])  (D-major so GEMM B-tiles are K-contiguous)
__global__ __launch_bounds__(256) void init_gt_kernel(const float* __restrict__ X,
                                                      bf16* __restrict__ gt) {
  const int n = blockIdx.x * 256 + threadIdx.x;
  const int d = blockIdx.y;
  const int b = blockIdx.z;
  gt[((long)b * D_ + d) * N_ + n] = (bf16)X[((long)b * N_ + n) * D_ + d];
}

// ---------------------------------------------------------------- diffusion GEMM
// out[n][d] = alpha * sum_k adj[n][k] * g[k][d] + (1-alpha) * X[n][d]
__global__ __launch_bounds__(256) void diffusion_step_kernel(
    const bf16* __restrict__ adjb,    // [B][N][N] bf16 row-major
    const bf16* __restrict__ gin_t,   // [B][D][N] bf16 (k contiguous)
    const float* __restrict__ X,      // [B][N][D] f32
    const float* __restrict__ alpha_logit,
    bf16* __restrict__ gout_t,        // [B][D][N]
    float* __restrict__ XsF,          // [B][N][D] (last step only)
    bf16*  __restrict__ XsB,          // [B][N][D] (last step only)
    int last)
{
  __shared__ bf16 As[2][BM * LDK];
  __shared__ bf16 Bs[2][BN * LDK];

  const int tid  = threadIdx.x;
  const int b    = blockIdx.z;
  const int mr   = blockIdx.x * BM;       // row base (N dim)
  const int nc   = blockIdx.y * BN;       // col base (D dim)
  const int wave = tid >> 5, lane = tid & 31;
  const int wm   = (wave & 3) * 32;       // wave tile M offset
  const int wn   = (wave >> 2) * 64;      // wave tile N offset
  const int rl   = lane & 15;
  const int ka   = (lane >> 4) * 8;       // A-frag / C-frag half select
  const int kb   = (lane >> 4) * 8;       // B-frag half select

  const bf16* Ag = adjb  + ((long)b * N_ + mr) * (long)N_;  // ld = N_
  const bf16* Bg = gin_t + ((long)b * D_ + nc) * (long)N_;  // ld = N_ (k along n)

  v8f zero;
  #pragma unroll
  for (int i = 0; i < 8; ++i) zero[i] = 0.0f;
  v8f acc[2][4];
  #pragma unroll
  for (int mi = 0; mi < 2; ++mi)
    #pragma unroll
    for (int ni = 0; ni < 4; ++ni) acc[mi][ni] = zero;

  const int lr = tid >> 1;           // 0..127: staged row
  const int lk = (tid & 1) * 16;     // k sub-chunk

  STAGE_ASYNC(As[0], Bs[0], 0, N_);

  #pragma unroll 1
  for (int kt = 0; kt < N_; kt += 2 * KT) {
    WAIT_ASYNC();
    __syncthreads();
    STAGE_ASYNC(As[1], Bs[1], kt + KT, N_);     // overlap with buf0 compute
    FRAG_WMMA(As[0], Bs[0]);
    __syncthreads();

    WAIT_ASYNC();
    __syncthreads();
    if (kt + 2 * KT < N_)
      STAGE_ASYNC(As[0], Bs[0], kt + 2 * KT, N_);
    FRAG_WMMA(As[1], Bs[1]);
    __syncthreads();
  }

  const float lg    = alpha_logit[0];
  const float alpha = 1.0f / (1.0f + __expf(-lg));
  const float beta  = 1.0f - alpha;

  #pragma unroll
  for (int mi = 0; mi < 2; ++mi) {
    #pragma unroll
    for (int ni = 0; ni < 4; ++ni) {
      const int d  = nc + wn + ni * 16 + rl;
      const int n0 = mr + wm + mi * 16 + ka;   // 8 consecutive rows per lane
      v8bf pk; float ov[8];
      #pragma unroll
      for (int v = 0; v < 8; ++v) {
        const float xv = X[((long)b * N_ + n0 + v) * D_ + d];
        const float o  = alpha * acc[mi][ni][v] + beta * xv;
        ov[v] = o; pk[v] = (bf16)o;
      }
      *reinterpret_cast<v8bf*>(&gout_t[((long)b * D_ + d) * N_ + n0]) = pk;
      if (last) {
        #pragma unroll
        for (int v = 0; v < 8; ++v) {
          XsF[((long)b * N_ + n0 + v) * D_ + d] = ov[v];
          XsB[((long)b * N_ + n0 + v) * D_ + d] = (bf16)ov[v];
        }
      }
    }
  }
}

// ---------------------------------------------------------------- proj1 + GELU
// H[b][n][a] = gelu_erf( sum_d Xs[b][n][d] * W1[a][d] + b1[a] )
__global__ __launch_bounds__(256) void proj_gelu_kernel(
    const bf16* __restrict__ XsB,    // [B][N][D]
    const bf16* __restrict__ W1b,    // [A][D]
    const float* __restrict__ b1,
    float* __restrict__ H)           // [B][N][A]
{
  __shared__ bf16 As[2][BM * LDK];
  __shared__ bf16 Bs[2][BN * LDK];

  const int tid  = threadIdx.x;
  const int b    = blockIdx.z;
  const int mr   = blockIdx.x * BM;
  const int wave = tid >> 5, lane = tid & 31;
  const int wm   = (wave & 3) * 32;
  const int wn   = (wave >> 2) * 64;
  const int rl   = lane & 15;
  const int ka   = (lane >> 4) * 8;
  const int kb   = (lane >> 4) * 8;

  const bf16* Ag = XsB + (long)b * N_ * D_ + (long)mr * D_;  // ld = D_
  const bf16* Bg = W1b;                                      // ld = D_, 128 rows

  v8f zero;
  #pragma unroll
  for (int i = 0; i < 8; ++i) zero[i] = 0.0f;
  v8f acc[2][4];
  #pragma unroll
  for (int mi = 0; mi < 2; ++mi)
    #pragma unroll
    for (int ni = 0; ni < 4; ++ni) acc[mi][ni] = zero;

  const int lr = tid >> 1;
  const int lk = (tid & 1) * 16;

  STAGE_ASYNC(As[0], Bs[0], 0, D_);

  #pragma unroll 1
  for (int kt = 0; kt < D_; kt += 2 * KT) {
    WAIT_ASYNC();
    __syncthreads();
    STAGE_ASYNC(As[1], Bs[1], kt + KT, D_);
    FRAG_WMMA(As[0], Bs[0]);
    __syncthreads();

    WAIT_ASYNC();
    __syncthreads();
    if (kt + 2 * KT < D_)
      STAGE_ASYNC(As[0], Bs[0], kt + 2 * KT, D_);
    FRAG_WMMA(As[1], Bs[1]);
    __syncthreads();
  }

  #pragma unroll
  for (int mi = 0; mi < 2; ++mi) {
    #pragma unroll
    for (int ni = 0; ni < 4; ++ni) {
      const int a  = wn + ni * 16 + rl;        // 0..127 (BN == A_)
      const int n0 = mr + wm + mi * 16 + ka;
      #pragma unroll
      for (int v = 0; v < 8; ++v) {
        const float h = acc[mi][ni][v] + b1[a];
        const float g = 0.5f * h * (1.0f + erff(h * 0.70710678118654752f));
        H[((long)b * N_ + n0 + v) * A_ + a] = g;
      }
    }
  }
}

// ---------------------------------------------------------------- proj2 (scores)
__global__ __launch_bounds__(256) void score_kernel(const float* __restrict__ H,
                                                    const float* __restrict__ w2,
                                                    float* __restrict__ f) {
  const int wave = threadIdx.x >> 5, lane = threadIdx.x & 31;
  const int row  = blockIdx.x * 8 + wave;       // (b,n) flat, 0..B_*N_-1
  const float* hr = H + (long)row * A_;
  float sum = 0.0f;
  for (int a = lane; a < A_; a += 32) sum += hr[a] * w2[a];
  #pragma unroll
  for (int off = 16; off > 0; off >>= 1) sum += __shfl_down(sum, off);
  if (lane == 0) f[row] = sum;
}

// ---------------------------------------------------------------- masked softmax over n
__global__ __launch_bounds__(1024) void softmax_kernel(const float* __restrict__ f,
                                                       const float* __restrict__ mask,
                                                       float* __restrict__ s) {
  __shared__ float red[1024];
  const int b = blockIdx.x, t = threadIdx.x;
  const float NEG = -__builtin_inff();
  float v0 = (mask[b * N_ + t]        > 0.0f) ? f[b * N_ + t]        : NEG;
  float v1 = (mask[b * N_ + t + 1024] > 0.0f) ? f[b * N_ + t + 1024] : NEG;
  red[t] = fmaxf(v0, v1);
  __syncthreads();
  for (int off = 512; off > 0; off >>= 1) {
    if (t < off) red[t] = fmaxf(red[t], red[t + off]);
    __syncthreads();
  }
  const float mx = red[0];
  __syncthreads();
  const float e0 = (v0 == NEG) ? 0.0f : __expf(v0 - mx);
  const float e1 = (v1 == NEG) ? 0.0f : __expf(v1 - mx);
  red[t] = e0 + e1;
  __syncthreads();
  for (int off = 512; off > 0; off >>= 1) {
    if (t < off) red[t] += red[t + off];
    __syncthreads();
  }
  const float inv = 1.0f / red[0];
  s[b * N_ + t]        = e0 * inv;
  s[b * N_ + t + 1024] = e1 * inv;
}

// ---------------------------------------------------------------- weighted pool
__global__ __launch_bounds__(256) void pool_kernel(const float* __restrict__ XsF,
                                                   const float* __restrict__ s,
                                                   float* __restrict__ z) {
  const int d = blockIdx.x * 256 + threadIdx.x;
  const int b = blockIdx.y;
  const float* xp = XsF + (long)b * N_ * D_ + d;
  const float* sp = s + b * N_;
  float sum = 0.0f;
  for (int n = 0; n < N_; ++n) sum += xp[(long)n * D_] * sp[n];
  z[b * D_ + d] = sum;
}

// ---------------------------------------------------------------- launcher
extern "C" void kernel_launch(void* const* d_in, const int* in_sizes, int n_in,
                              void* d_out, int out_size, void* d_ws, size_t ws_size,
                              hipStream_t stream) {
  (void)in_sizes; (void)n_in; (void)out_size; (void)ws_size;
  const float* X    = (const float*)d_in[0];
  const float* adj  = (const float*)d_in[1];
  const float* mask = (const float*)d_in[2];
  const float* W1   = (const float*)d_in[3];
  const float* b1   = (const float*)d_in[4];
  const float* w2   = (const float*)d_in[5];
  const float* alpha_logit = (const float*)d_in[6];
  float* z = (float*)d_out;

  char* ws = (char*)d_ws;
  size_t off = 0;
  auto alloc = [&](size_t bytes) -> void* {
    void* p = ws + off;
    off += (bytes + 255) & ~(size_t)255;
    return p;
  };
  bf16*  adjb = (bf16*) alloc((size_t)B_ * N_ * N_ * 2);   // 67 MB, L2-resident
  bf16*  g0   = (bf16*) alloc((size_t)B_ * D_ * N_ * 2);
  bf16*  g1   = (bf16*) alloc((size_t)B_ * D_ * N_ * 2);
  float* XsF  = (float*)alloc((size_t)B_ * N_ * D_ * 4);
  bf16*  XsB  = (bf16*) alloc((size_t)B_ * N_ * D_ * 2);
  bf16*  W1b  = (bf16*) alloc((size_t)A_ * D_ * 2);
  float* H    = (float*)alloc((size_t)B_ * N_ * A_ * 4);
  float* f    = (float*)alloc((size_t)B_ * N_ * 4);
  float* s    = (float*)alloc((size_t)B_ * N_ * 4);

  { long n4 = (long)B_ * N_ * N_ / 4;
    cvt4_kernel<<<dim3((unsigned)((n4 + 255) / 256)), dim3(256), 0, stream>>>(adj, adjb, n4); }
  { long n4 = (long)A_ * D_ / 4;
    cvt4_kernel<<<dim3((unsigned)((n4 + 255) / 256)), dim3(256), 0, stream>>>(W1, W1b, n4); }
  init_gt_kernel<<<dim3(N_ / 256, D_, B_), dim3(256), 0, stream>>>(X, g0);

  bf16* gin = g0; bf16* gout = g1;
  for (int step = 0; step < 10; ++step) {
    const int last = (step == 9) ? 1 : 0;
    diffusion_step_kernel<<<dim3(N_ / BM, D_ / BN, B_), dim3(256), 0, stream>>>(
        adjb, gin, X, alpha_logit, gout, XsF, XsB, last);
    bf16* t = gin; gin = gout; gout = t;
  }

  proj_gelu_kernel<<<dim3(N_ / BM, 1, B_), dim3(256), 0, stream>>>(XsB, W1b, b1, H);
  score_kernel<<<dim3(B_ * N_ / 8), dim3(256), 0, stream>>>(H, w2, f);
  softmax_kernel<<<dim3(B_), dim3(1024), 0, stream>>>(f, mask, s);
  pool_kernel<<<dim3(D_ / 256, B_), dim3(256), 0, stream>>>(XsF, s, z);
}
